// LIF_no_grad_31516470018622
// MI455X (gfx1250) — matrix operand order repaired and checked by reading, not compiled, not run on hardware
//
#include <hip/hip_runtime.h>
#include <hip/hip_bf16.h>

// ---------------------------------------------------------------------------
// LIF recurrent network, T=2048 sequential steps of:
//   I = s @ W_eff + 1.75*x[t]   (32x2048 @ 2048x2048, bf16 WMMA, f32 accum)
//   elementwise LIF update of v, s; out = sigmoid(v_next - 30)
//
// v5: WGP-scope B prefetch via inline asm using the immediate offset field
//     (shares the load's base VGPR pair -> no per-iteration 64-bit adds) and
//     no "memory" clobber (prefetch has no compiler-visible side effect), so
//     the scheduler can pipeline around it like v3.
// ---------------------------------------------------------------------------

typedef __attribute__((ext_vector_type(16))) __bf16 v16bf;
typedef __attribute__((ext_vector_type(8)))  __bf16 v8bf;
typedef __attribute__((ext_vector_type(8)))  float  v8f;

#define T_STEPS 2048
#define BATCH   32
#define NN      2048          // neurons
#define NTILES  (NN / 16)     // 128 column tiles
#define KTILES  (NN / 32)     // 64 k-steps of 32
#define WAVES_PER_BLOCK 8
#define BLOCKS  (NTILES / WAVES_PER_BLOCK)   // 16
#define LDS_BYTES (BATCH * NN * 2)           // 128 KB staged A matrix

// WGP-scope speculative prefetch 4 k-tiles (4KB) ahead of the given address.
// Scope 0 (no scope token) = fill all cache levels, stop on WGP hit. No
// memory clobber: prefetch has no architecturally visible effect, so the
// scheduler may move it freely; volatile keeps it from being deleted.
__device__ __forceinline__ void prefetch_wgp_4k(const void* p) {
    asm volatile("global_prefetch_b8 %0, off offset:4096"
                 :: "v"((unsigned long long)(uintptr_t)p));
}

// ---------------------------------------------------------------------------
// One-time: build Wp[ntile][ktile][lane][16] bf16 fragments.
// Fragment element t of lane L in tile (ntile, kt) is
//   W_eff[i][j],  j = ntile*16 + (L&15),  i = kt*32 + (L>=16 ? 8 : 0) + (t&7) + (t>=8 ? 16 : 0)
// W_eff[i][j] = w[i][j] * neuron_types[i], diagonal zeroed.
// ---------------------------------------------------------------------------
__global__ void lif_prep_w(const float* __restrict__ w,
                           const int* __restrict__ ntypes,
                           __bf16* __restrict__ Wp) {
    int gid = blockIdx.x * blockDim.x + threadIdx.x;   // 0 .. NN*NN-1 (4M)
    if (gid >= NN * NN) return;
    int t     = gid & 15;
    int L     = (gid >> 4) & 31;
    int kt    = (gid >> 9) & (KTILES - 1);
    int ntile = gid >> 15;
    int j = ntile * 16 + (L & 15);
    int i = kt * 32 + ((L >> 4) ? 8 : 0) + (t & 7) + ((t >> 3) ? 16 : 0);
    float val = (i == j) ? 0.0f : w[i * NN + j] * (float)ntypes[i];
    Wp[gid] = (__bf16)val;
}

// ---------------------------------------------------------------------------
// One-time: initialize state from v0, s0 (harness poisons d_ws before timing).
// ---------------------------------------------------------------------------
__global__ void lif_init_state(const float* __restrict__ v0,
                               const float* __restrict__ s0,
                               float* __restrict__ v, float* __restrict__ s,
                               __bf16* __restrict__ sb0, __bf16* __restrict__ sb1) {
    int gid = blockIdx.x * blockDim.x + threadIdx.x;
    if (gid >= BATCH * NN) return;
    v[gid] = v0[gid];
    float sv = s0[gid];
    s[gid] = sv;
    sb0[gid] = (__bf16)sv;
    sb1[gid] = (__bf16)sv;
}

// ---------------------------------------------------------------------------
// One timestep. Grid: 16 blocks x 256 threads (8 waves). The block stages the
// full A matrix (prev-step s, bf16, row-major 32x2048) into LDS; each wave
// then owns one 16-column tile of C (both 16x16 row-tiles -> two f32 accs).
// ---------------------------------------------------------------------------
__global__ __launch_bounds__(256)
void lif_step(const __bf16* __restrict__ Wp,
              const __bf16* __restrict__ s_in,   // bf16 A matrix (prev step)
              __bf16* __restrict__ s_out,        // bf16 A matrix (next step)
              float* __restrict__ v,
              float* __restrict__ s,
              const float* __restrict__ x_t,     // x_in + t*B*N
              const float* __restrict__ E_L,
              const float* __restrict__ tau_m,
              const float* __restrict__ tau_s,
              float* __restrict__ out_t) {       // out + t*B*N
    extern __shared__ __align__(16) char lds_raw[];
    __bf16* sA = (__bf16*)lds_raw;               // [BATCH][NN] row-major

    // ---- stage A into LDS: 8192 chunks of 16B, 32 per thread --------------
    {
        const uint4* gsrc = (const uint4*)s_in;
        uint4*       ldst = (uint4*)sA;
        #pragma unroll
        for (int i = 0; i < (BATCH * NN) / (8 * 256); ++i) {
            int c = threadIdx.x + 256 * i;
            ldst[c] = gsrc[c];
        }
    }
    __syncthreads();

    const int lane = threadIdx.x & 31;
    const int wave = threadIdx.x >> 5;
    const int gw   = blockIdx.x * WAVES_PER_BLOCK + wave;  // 0..127: col tile
    const int col  = lane & 15;
    const int half = lane >> 4;                  // 0 or 1
    const int n    = gw * 16 + col;

    v8f acc0 = {};   // rows 0..15
    v8f acc1 = {};   // rows 16..31

    // A-fragment row bases in LDS (ISA 16-bit A layout: lane holds row
    // m = lane&15; lanes 0-15 cover K [0..7] & [16..23], lanes 16-31
    // cover K [8..15] & [24..31]).
    const __bf16* a0base = sA + (size_t)(col) * NN;        // m = col
    const __bf16* a1base = sA + (size_t)(16 + col) * NN;   // m = 16 + col
    const int aoff = half * 8;

    const __bf16* bbase = Wp + ((size_t)gw * KTILES) * 512 + (size_t)lane * 16;

    for (int k0 = 0; k0 < NN; k0 += 32) {
        // B fragment: 16 contiguous bf16 per lane (pre-swizzled, 32B aligned)
        const __bf16* bp = bbase + (size_t)(k0 >> 5) * 512;
        v16bf bfrag = *(const v16bf*)bp;
        // WGP-scope prefetch 4 k-tiles ahead (immediate offset; overrun past
        // this wave's slab just touches the next slab -- harmless).
        prefetch_wgp_4k(bp);

        // A fragments from LDS: two aligned 16B chunks per lane per row-tile
        v8bf a0lo = *(const v8bf*)(a0base + k0 + aoff);
        v8bf a0hi = *(const v8bf*)(a0base + k0 + 16 + aoff);
        v8bf a1lo = *(const v8bf*)(a1base + k0 + aoff);
        v8bf a1hi = *(const v8bf*)(a1base + k0 + 16 + aoff);
        v16bf afrag0 = __builtin_shufflevector(a0lo, a0hi,
                          0,1,2,3,4,5,6,7,8,9,10,11,12,13,14,15);
        v16bf afrag1 = __builtin_shufflevector(a1lo, a1hi,
                          0,1,2,3,4,5,6,7,8,9,10,11,12,13,14,15);

        acc0 = __builtin_amdgcn_wmma_f32_16x16x32_bf16(
                   false, afrag0, false, bfrag, (short)0, acc0, false, false);
        acc1 = __builtin_amdgcn_wmma_f32_16x16x32_bf16(
                   false, afrag1, false, bfrag, (short)0, acc1, false, false);
    }

    // Per-neuron constants (uniform per lane: n fixed across all 16 elements)
    const float eL      = E_L[n];
    const float inv_tm  = 1.0f / tau_m[n];
    const float inv_ts  = 1.0f / tau_s[n];
    const float dv_max  = 30.0f - eL;            // THRESH - E_L
    const float norm_R  = dv_max * 1.1f;         // (THRESH - E_L) * R_CONST
    const float inv_dvm = 1.0f / dv_max;

    // C layout: VGPR r holds row (r + 8*half); col = lane&15.
    #pragma unroll
    for (int r = 0; r < 8; ++r) {
        #pragma unroll
        for (int mt = 0; mt < 2; ++mt) {
            const int   b   = mt * 16 + r + half * 8;
            const int   idx = b * NN + n;
            const float accv = (mt == 0) ? acc0[r] : acc1[r];
            const float I   = accv + 1.75f * x_t[idx];
            const float vc  = v[idx];
            const float sc  = s[idx];
            const float dv  = (eL - vc + I * norm_R) * inv_tm;
            const float vn  = vc + dv;
            const float gat = fminf(fmaxf(vn * (1.0f / 30.0f), 0.0f), 1.0f);
            const float dvr = fminf(fmaxf(dv * inv_dvm, 0.0f), 1.0f);
            const float sn  = sc + (gat * dvr - sc) * inv_ts;
            v[idx]     = (vn >= 30.0f) ? eL : vn;
            s[idx]     = sn;
            s_out[idx] = (__bf16)sn;
            out_t[idx] = 1.0f / (1.0f + __expf(30.0f - vn));
        }
    }
}

// ---------------------------------------------------------------------------
// Host-side orchestration: prep + init once, then T sequential step launches
// (inter-step dependency enforced by stream ordering; bf16 s double-buffered).
// ---------------------------------------------------------------------------
extern "C" void kernel_launch(void* const* d_in, const int* in_sizes, int n_in,
                              void* d_out, int out_size, void* d_ws, size_t ws_size,
                              hipStream_t stream) {
    const float* x      = (const float*)d_in[0];  // (T, B, N)
    const float* v0     = (const float*)d_in[1];  // (B, N)
    const float* s0     = (const float*)d_in[2];  // (B, N)
    const float* w      = (const float*)d_in[3];  // (N, N)
    const float* E_L    = (const float*)d_in[4];  // (N,)
    const float* tau_m  = (const float*)d_in[5];  // (N,)
    const float* tau_s  = (const float*)d_in[6];  // (N,)
    const int*   ntypes = (const int*)d_in[7];    // (N,)
    float*       out    = (float*)d_out;          // (T, B, N)

    char* ws = (char*)d_ws;
    __bf16* Wp  = (__bf16*)ws;                                   // 8 MB
    float*  v   = (float*)(ws + (size_t)NN * NN * sizeof(__bf16));
    float*  s   = v + (size_t)BATCH * NN;                        // 256 KB each
    __bf16* sb0 = (__bf16*)(s + (size_t)BATCH * NN);             // 128 KB
    __bf16* sb1 = sb0 + (size_t)BATCH * NN;                      // 128 KB

    lif_prep_w<<<(NN * NN) / 256, 256, 0, stream>>>(w, ntypes, Wp);
    lif_init_state<<<(BATCH * NN) / 256, 256, 0, stream>>>(v0, s0, v, s, sb0, sb1);

    for (int t = 0; t < T_STEPS; ++t) {
        const __bf16* sin  = (t & 1) ? sb1 : sb0;
        __bf16*       sout = (t & 1) ? sb0 : sb1;
        lif_step<<<BLOCKS, 256, LDS_BYTES, stream>>>(
            Wp, sin, sout, v, s,
            x + (size_t)t * BATCH * NN,
            E_L, tau_m, tau_s,
            out + (size_t)t * BATCH * NN);
    }
}
